// LoopyBP_21002390078198
// MI455X (gfx1250) — matrix operand order repaired and checked by reading, not compiled, not run on hardware
//
#include <hip/hip_runtime.h>
#include <cstddef>
#include <cstdint>

#define NUM_ATOMS 100000
#define NUM_BONDS 200000
#define MAX_NB 6
#define ATOM_FDIM 133
#define BOND_FDIM 147
#define HIDDEN 300
#define READOUT 512
#define DEPTH 6
#define ATOMS_PER_MOL 25
#define NUM_MOLS (NUM_ATOMS / ATOMS_PER_MOL)
#define MSG_LD 304  // bf16 message row stride (300 padded -> 608B, 16B aligned)

typedef __attribute__((ext_vector_type(16))) __bf16 v16bf;
typedef __attribute__((ext_vector_type(8)))  float  v8f;

union AFrag { v16bf v; __bf16 e[16]; };
union BFrag { v16bf v; uint4 q[2]; };
union BfBits { __bf16 b; unsigned short u; };
union F16x  { float4 q[4]; float f[16]; };
union BF8x  { uint4 q; unsigned short s[8]; };

__device__ __forceinline__ float bf2f(unsigned short u) {
    union { unsigned int i; float f; } c;
    c.i = ((unsigned int)u) << 16;
    return c.f;
}

// ---------------------------------------------------------------------------
// Pack fp32 weight W[K][N] (row-major, y = x@W) into per-lane WMMA B-fragment
// layout, bf16, zero padded. ntilesP is a multiple of 4 so the GEMM LDS
// staging is unconditional.
//   tile (kt,nt): 32 lanes x 16 bf16 (32B per lane, contiguous).
//   lanes 0-15:  N = nt*16+lane,    K = kt*32 + i (i<8), kt*32 + 8 + i (i>=8)
//   lanes 16-31: N = nt*16+lane-16, K = kt*32 + 8 + i (i<8), kt*32 + 16 + i (i>=8)
// ---------------------------------------------------------------------------
__global__ void pack_weights(const float* __restrict__ W, int K, int N,
                             unsigned short* __restrict__ Wp,
                             int ktiles, int ntilesP) {
    int t = blockIdx.x * blockDim.x + threadIdx.x;
    int total = ktiles * ntilesP * 32;
    if (t >= total) return;
    int lane = t & 31;
    int tile = t >> 5;
    int nt = tile % ntilesP;
    int kt = tile / ntilesP;
    int nn = nt * 16 + (lane & 15);
    int kb = kt * 32 + ((lane & 16) ? 8 : 0);
    unsigned short* dst = Wp + (size_t)t * 16;
#pragma unroll
    for (int i = 0; i < 16; ++i) {
        int k = kb + (i < 8 ? i : i + 8);
        float v = (k < K && nn < N) ? W[(size_t)k * N + nn] : 0.0f;
        BfBits cv; cv.b = (__bf16)v;
        dst[i] = cv.u;
    }
}

// ---------------------------------------------------------------------------
// a_message[a][h] = sum_{j<6} message[a2b[a][j]][h]  (bf16 msg in, fp32 out)
// One thread = 4 channels (ushort4 load / float4 store).
// ---------------------------------------------------------------------------
__global__ void atom_aggregate(const unsigned short* __restrict__ msgb,
                               const int* __restrict__ a2b,
                               float* __restrict__ aMsg) {
    const int H4 = HIDDEN / 4;  // 75
    int t = blockIdx.x * blockDim.x + threadIdx.x;
    if (t >= NUM_ATOMS * H4) return;
    int a = t / H4;
    int c = t - a * H4;
    float s0 = 0.f, s1 = 0.f, s2 = 0.f, s3 = 0.f;
#pragma unroll
    for (int j = 0; j < MAX_NB; ++j) {
        int b = a2b[a * MAX_NB + j];
        ushort4 v = *(const ushort4*)(msgb + (size_t)b * MSG_LD + c * 4);
        s0 += bf2f(v.x); s1 += bf2f(v.y); s2 += bf2f(v.z); s3 += bf2f(v.w);
    }
    *(float4*)(aMsg + (size_t)a * HIDDEN + c * 4) = make_float4(s0, s1, s2, s3);
}

// ---------------------------------------------------------------------------
// mol_vecs[m][h4] = mean over 25 contiguous atoms of atom_hiddens (fp32)
// ---------------------------------------------------------------------------
__global__ void mol_pool(const float* __restrict__ atomH,
                         float* __restrict__ molv) {
    const int H4 = HIDDEN / 4;
    int t = blockIdx.x * blockDim.x + threadIdx.x;
    if (t >= NUM_MOLS * H4) return;
    int m = t / H4;
    int c = t - m * H4;
    float4 s = make_float4(0.f, 0.f, 0.f, 0.f);
#pragma unroll 5
    for (int i = 0; i < ATOMS_PER_MOL; ++i) {
        float4 v = *(const float4*)(atomH + (size_t)(m * ATOMS_PER_MOL + i) * HIDDEN + c * 4);
        s.x += v.x; s.y += v.y; s.z += v.z; s.w += v.w;
    }
    const float inv = 1.0f / (float)ATOMS_PER_MOL;
    s.x *= inv; s.y *= inv; s.z *= inv; s.w *= inv;
    *(float4*)(molv + (size_t)m * HIDDEN + c * 4) = s;
}

// ---------------------------------------------------------------------------
// out[m] = b_last + dot(h1[m], W_last)   (exact fp32, tiny)
// ---------------------------------------------------------------------------
__global__ void final_dot(const float* __restrict__ h1,
                          const float* __restrict__ Wl,
                          const float* __restrict__ bl,
                          float* __restrict__ out) {
    int m = blockIdx.x * blockDim.x + threadIdx.x;
    if (m >= NUM_MOLS) return;
    float s = bl[0];
    const float4* row = (const float4*)(h1 + (size_t)m * READOUT);
    const float4* w   = (const float4*)Wl;
#pragma unroll 4
    for (int k = 0; k < READOUT / 4; ++k) {
        float4 a = row[k], b = w[k];
        s += a.x * b.x + a.y * b.y + a.z * b.z + a.w * b.w;
    }
    out[m] = s;
}

// ---------------------------------------------------------------------------
// A-fragment loaders (ISA 16-bit A 16x32 layout: elements 0-7 <-> K kA..kA+7,
// elements 8-15 <-> K kA+16..kA+23; kA already includes the half-wave shift).
// Full tiles: branch-free wide loads. Tail: clamp + cndmask, branch-free.
// ---------------------------------------------------------------------------
template <int MODE, bool ALIGNED>
__device__ __forceinline__ AFrag loadA_full(const float* pX,
                                            const float* pA,
                                            const unsigned short* pR,
                                            const float* pF, const float* pM,
                                            int kA) {
    AFrag af;
    if (MODE == 1) {
        // aMsg fp32 (float4 x4) minus bf16 message (uint4 x2)
        const float4* qa = (const float4*)(pA + kA);
        F16x ta;
        ta.q[0] = qa[0]; ta.q[1] = qa[1]; ta.q[2] = qa[4]; ta.q[3] = qa[5];
        BF8x r0, r1;
        r0.q = *(const uint4*)(pR + kA);
        r1.q = *(const uint4*)(pR + kA + 16);
#pragma unroll
        for (int i = 0; i < 8; ++i) {
            af.e[i]     = (__bf16)(ta.f[i]     - bf2f(r0.s[i]));
            af.e[8 + i] = (__bf16)(ta.f[8 + i] - bf2f(r1.s[i]));
        }
    } else if (MODE == 0 && ALIGNED) {
        const float4* q = (const float4*)(pX + kA);
        F16x t;
        t.q[0] = q[0]; t.q[1] = q[1]; t.q[2] = q[4]; t.q[3] = q[5];
#pragma unroll
        for (int i = 0; i < 16; ++i) af.e[i] = (__bf16)t.f[i];
    } else if (MODE == 0) {
#pragma unroll
        for (int i = 0; i < 8; ++i) {
            af.e[i]     = (__bf16)pX[kA + i];
            af.e[8 + i] = (__bf16)pX[kA + 16 + i];
        }
    } else {  // MODE 2: branch-free pointer select between f_atoms and aMsg
#pragma unroll
        for (int i = 0; i < 8; ++i) {
            int k0 = kA + i, k1 = kA + 16 + i;
            const float* p0 = (k0 < ATOM_FDIM) ? (pF + k0) : (pM + (k0 - ATOM_FDIM));
            const float* p1 = (k1 < ATOM_FDIM) ? (pF + k1) : (pM + (k1 - ATOM_FDIM));
            af.e[i]     = (__bf16)*p0;
            af.e[8 + i] = (__bf16)*p1;
        }
    }
    return af;
}

template <int MODE>
__device__ __forceinline__ AFrag loadA_tail(const float* pX,
                                            const float* pA,
                                            const unsigned short* pR,
                                            const float* pF, const float* pM,
                                            int kA, int K) {
    AFrag af;
#pragma unroll
    for (int i = 0; i < 8; ++i) {
#pragma unroll
        for (int s = 0; s < 2; ++s) {
            int k = kA + s * 16 + i;
            bool ok = k < K;
            int kc = ok ? k : 0;
            float v;
            if (MODE == 0) {
                v = pX[kc];
            } else if (MODE == 1) {
                v = pA[kc] - bf2f(pR[kc]);
            } else {
                const float* p = (kc < ATOM_FDIM) ? (pF + kc) : (pM + (kc - ATOM_FDIM));
                v = *p;
            }
            v = ok ? v : 0.f;
            af.e[s * 8 + i] = (__bf16)v;
        }
    }
    return af;
}

// ---------------------------------------------------------------------------
// WMMA bf16 GEMM: epilogue(A_src @ Wp + bias)
//   MODE 0: A[m][k] = X[m*ldx + k]          (ALIGNED: ldx%4==0)
//   MODE 1: A[m][k] = aMsg[b2a[m]][k] - bf16 msg[b2revb[m]][k] (fused gather)
//   MODE 2: A[m][k] = k<133 ? fAtoms[m][k] : aMsg[m][k-133]    (fused concat)
// Stores: OUTF32 -> out[m*N+n] = v (post-ReLU if RELU);
//         OUTBF16 -> outb[m*ldOutB+n] = bf16(max(v,0))   (message buffers)
// Wave = 16Mx64N; all 4 B fragments preloaded so the 4
// v_wmma_f32_16x16x32_bf16 issue back-to-back per K-tile; block = 8 waves.
// ---------------------------------------------------------------------------
template <int MODE, bool ALIGNED, bool RESID, bool OUTF32, bool OUTBF16, bool RELU>
__global__ __launch_bounds__(256) void gemm_wmma(
    const float* __restrict__ X, int ldx,
    const float* __restrict__ aMsg, const unsigned short* __restrict__ msgb,
    const int* __restrict__ b2a, const int* __restrict__ b2revb,
    const float* __restrict__ fAtoms,
    const unsigned short* __restrict__ Wp, int ntilesP,
    const float* __restrict__ bias, const float* __restrict__ resid,
    float* __restrict__ out, unsigned short* __restrict__ outb, int ldOutB,
    int M, int N, int K) {
    __shared__ uint4 sB[256];  // 4 B-tiles x 1KB

    const int tid    = threadIdx.x;
    const int lane   = tid & 31;
    const int wave   = tid >> 5;
    const int mTile  = blockIdx.x * 128 + wave * 16;
    const int nBase  = blockIdx.y * 64;
    const int ntBase = blockIdx.y * 4;

    const int rowRaw = mTile + (lane & 15);
    const int row = (rowRaw < M) ? rowRaw : (M - 1);

    const float* pX = nullptr;
    const float* pA = nullptr;
    const unsigned short* pR = nullptr;
    const float* pF = nullptr;
    const float* pM = nullptr;
    if (MODE == 0) pX = X + (size_t)row * ldx;
    if (MODE == 1) {
        pA = aMsg + (size_t)b2a[row] * HIDDEN;
        pR = msgb + (size_t)b2revb[row] * MSG_LD;
    }
    if (MODE == 2) {
        pF = fAtoms + (size_t)row * ATOM_FDIM;
        pM = aMsg + (size_t)row * HIDDEN;
    }

    v8f acc[4];
#pragma unroll
    for (int j = 0; j < 4; ++j)
        acc[j] = (v8f){0.f, 0.f, 0.f, 0.f, 0.f, 0.f, 0.f, 0.f};

    const int kHalf = (lane & 16) ? 8 : 0;  // A-layout half-wave shift
    const int copyJ = tid >> 6;             // B staging: thread -> tile,word
    const int copyW = tid & 63;

    const int kfull = K >> 5;
    const bool hasTail = (K & 31) != 0;

    for (int kt = 0; kt < kfull; ++kt) {
        const int kA = kt * 32 + kHalf;
        // Prefetch next K-tile of the gathered/streamed A rows (speculative).
        if (MODE == 1) {
            __builtin_prefetch(pA + kA + 32, 0, 0);
            __builtin_prefetch(pR + kA + 32, 0, 0);
        } else if (MODE == 0 && ALIGNED) {
            __builtin_prefetch(pX + kA + 32, 0, 0);
        }

        __syncthreads();
        sB[tid] = ((const uint4*)Wp)[((size_t)kt * ntilesP + ntBase + copyJ) * 64 + copyW];
        __syncthreads();

        AFrag af = loadA_full<MODE, ALIGNED>(pX, pA, pR, pF, pM, kA);
        BFrag bf[4];
#pragma unroll
        for (int j = 0; j < 4; ++j) {
            bf[j].q[0] = sB[j * 64 + lane * 2];
            bf[j].q[1] = sB[j * 64 + lane * 2 + 1];
        }
#pragma unroll
        for (int j = 0; j < 4; ++j)
            acc[j] = __builtin_amdgcn_wmma_f32_16x16x32_bf16(
                false, af.v, false, bf[j].v, (short)0, acc[j], false, false);
    }
    if (hasTail) {
        __syncthreads();
        sB[tid] = ((const uint4*)Wp)[((size_t)kfull * ntilesP + ntBase + copyJ) * 64 + copyW];
        __syncthreads();

        AFrag af = loadA_tail<MODE>(pX, pA, pR, pF, pM, kfull * 32 + kHalf, K);
        BFrag bf[4];
#pragma unroll
        for (int j = 0; j < 4; ++j) {
            bf[j].q[0] = sB[j * 64 + lane * 2];
            bf[j].q[1] = sB[j * 64 + lane * 2 + 1];
        }
#pragma unroll
        for (int j = 0; j < 4; ++j)
            acc[j] = __builtin_amdgcn_wmma_f32_16x16x32_bf16(
                false, af.v, false, bf[j].v, (short)0, acc[j], false, false);
    }

    // Epilogue (D layout: lane%16 = N, VGPR r -> M = r (+8 for upper half-wave))
    const int rHalf = (lane & 16) ? 8 : 0;
#pragma unroll
    for (int j = 0; j < 4; ++j) {
        const int col = nBase + j * 16 + (lane & 15);
        if (col >= N) continue;
        const float bv = bias[col];
#pragma unroll
        for (int r = 0; r < 8; ++r) {
            const int orow = mTile + rHalf + r;
            if (orow >= M) continue;
            float v = acc[j][r] + bv;
            if (RESID) v += resid[(size_t)orow * N + col];
            if (RELU) v = fmaxf(v, 0.f);
            if (OUTF32) out[(size_t)orow * N + col] = v;
            if (OUTBF16) {
                BfBits cv; cv.b = (__bf16)fmaxf(v, 0.f);
                outb[(size_t)orow * ldOutB + col] = cv.u;
            }
        }
    }
}

// ---------------------------------------------------------------------------
extern "C" void kernel_launch(void* const* d_in, const int* in_sizes, int n_in,
                              void* d_out, int out_size, void* d_ws, size_t ws_size,
                              hipStream_t stream) {
    const float* f_atoms = (const float*)d_in[0];
    const float* f_bonds = (const float*)d_in[1];
    const int* a2b = (const int*)d_in[2];
    const int* b2a = (const int*)d_in[3];
    const int* b2revb = (const int*)d_in[4];
    // d_in[5] = mol_id (unused: mols are 25 contiguous atoms by construction)
    const float* W_i = (const float*)d_in[6];
    const float* b_i = (const float*)d_in[7];
    const float* W_h = (const float*)d_in[8];
    const float* b_h = (const float*)d_in[9];
    const float* W_o = (const float*)d_in[10];
    const float* b_o = (const float*)d_in[11];
    const float* W_fc0 = (const float*)d_in[12];
    const float* b_fc0 = (const float*)d_in[13];
    const float* W_fc1 = (const float*)d_in[14];
    const float* b_fc1 = (const float*)d_in[15];
    const float* W_last = (const float*)d_in[16];
    const float* b_last = (const float*)d_in[17];
    float* outp = (float*)d_out;

    char* ws = (char*)d_ws;
    size_t off = 0;
    auto carve = [&](size_t bytes) -> char* {
        char* p = ws + off;
        off += (bytes + 255) & ~(size_t)255;
        return p;
    };
    float* inp = (float*)carve((size_t)NUM_BONDS * HIDDEN * 4);             // 240 MB fp32
    unsigned short* msgA = (unsigned short*)carve((size_t)NUM_BONDS * MSG_LD * 2);  // 121.6 MB bf16
    unsigned short* msgB = (unsigned short*)carve((size_t)NUM_BONDS * MSG_LD * 2);  // 121.6 MB bf16
    float* aMsg = (float*)carve((size_t)NUM_ATOMS * HIDDEN * 4);            // 120 MB fp32
    float* atomH = (float*)carve((size_t)NUM_ATOMS * HIDDEN * 4);           // 120 MB fp32
    float* molv = (float*)carve((size_t)NUM_MOLS * HIDDEN * 4);
    float* h0   = (float*)carve((size_t)NUM_MOLS * READOUT * 4);
    float* h1   = (float*)carve((size_t)NUM_MOLS * READOUT * 4);

    auto ntP = [](int N) { return ((((N + 15) / 16) + 3) / 4) * 4; };
    auto packedBytes = [&](int K, int N) {
        return (size_t)((K + 31) / 32) * ntP(N) * 1024;
    };
    unsigned short* wpI  = (unsigned short*)carve(packedBytes(BOND_FDIM, HIDDEN));
    unsigned short* wpH  = (unsigned short*)carve(packedBytes(HIDDEN, HIDDEN));
    unsigned short* wpO  = (unsigned short*)carve(packedBytes(ATOM_FDIM + HIDDEN, HIDDEN));
    unsigned short* wpF0 = (unsigned short*)carve(packedBytes(HIDDEN, READOUT));
    unsigned short* wpF1 = (unsigned short*)carve(packedBytes(READOUT, READOUT));

    auto pack = [&](const float* W, int K, int N, unsigned short* Wp) {
        int kt = (K + 31) / 32, nt = ntP(N);
        int total = kt * nt * 32;
        pack_weights<<<(total + 255) / 256, 256, 0, stream>>>(W, K, N, Wp, kt, nt);
    };
    pack(W_i, BOND_FDIM, HIDDEN, wpI);
    pack(W_h, HIDDEN, HIDDEN, wpH);
    pack(W_o, ATOM_FDIM + HIDDEN, HIDDEN, wpO);
    pack(W_fc0, HIDDEN, READOUT, wpF0);
    pack(W_fc1, READOUT, READOUT, wpF1);

    auto gemmGrid = [](int M, int N) { return dim3((M + 127) / 128, (N + 63) / 64); };

    // inp = f_bonds @ W_i + b_i (fp32) ; msgA = bf16(relu(inp))
    gemm_wmma<0, false, false, true, true, false><<<gemmGrid(NUM_BONDS, HIDDEN), 256, 0, stream>>>(
        f_bonds, BOND_FDIM, nullptr, nullptr, nullptr, nullptr, nullptr,
        wpI, ntP(HIDDEN), b_i, nullptr, inp, msgA, MSG_LD, NUM_BONDS, HIDDEN, BOND_FDIM);

    const int aggBlocks = (NUM_ATOMS * (HIDDEN / 4) + 255) / 256;
    for (int d = 0; d < DEPTH - 1; ++d) {
        unsigned short* cur = (d & 1) ? msgB : msgA;
        unsigned short* nxt = (d & 1) ? msgA : msgB;
        atom_aggregate<<<aggBlocks, 256, 0, stream>>>(cur, a2b, aMsg);
        // nxt = bf16(relu(inp + (aMsg[b2a] - cur[b2revb]) @ W_h + b_h))
        gemm_wmma<1, true, true, false, true, true><<<gemmGrid(NUM_BONDS, HIDDEN), 256, 0, stream>>>(
            nullptr, 0, aMsg, cur, b2a, b2revb, nullptr,
            wpH, ntP(HIDDEN), b_h, inp, nullptr, nxt, MSG_LD, NUM_BONDS, HIDDEN, HIDDEN);
    }
    unsigned short* msgFinal = msgB;  // DEPTH-1 = 5 iterations: A->B->A->B->A->B
    atom_aggregate<<<aggBlocks, 256, 0, stream>>>(msgFinal, a2b, aMsg);

    // atom_hiddens = relu([f_atoms | aMsg] @ W_o + b_o)  (fp32)
    gemm_wmma<2, false, false, true, false, true><<<gemmGrid(NUM_ATOMS, HIDDEN), 256, 0, stream>>>(
        nullptr, 0, aMsg, nullptr, nullptr, nullptr, f_atoms,
        wpO, ntP(HIDDEN), b_o, nullptr, atomH, nullptr, 0,
        NUM_ATOMS, HIDDEN, ATOM_FDIM + HIDDEN);

    mol_pool<<<(NUM_MOLS * (HIDDEN / 4) + 255) / 256, 256, 0, stream>>>(atomH, molv);

    gemm_wmma<0, true, false, true, false, true><<<gemmGrid(NUM_MOLS, READOUT), 256, 0, stream>>>(
        molv, HIDDEN, nullptr, nullptr, nullptr, nullptr, nullptr,
        wpF0, ntP(READOUT), b_fc0, nullptr, h0, nullptr, 0, NUM_MOLS, READOUT, HIDDEN);

    gemm_wmma<0, true, false, true, false, true><<<gemmGrid(NUM_MOLS, READOUT), 256, 0, stream>>>(
        h0, READOUT, nullptr, nullptr, nullptr, nullptr, nullptr,
        wpF1, ntP(READOUT), b_fc1, nullptr, h1, nullptr, 0, NUM_MOLS, READOUT, READOUT);

    final_dot<<<(NUM_MOLS + 255) / 256, 256, 0, stream>>>(h1, W_last, b_last, outp);
}
